// Myattention_38706245271991
// MI455X (gfx1250) — matrix operand order repaired and checked by reading, not compiled, not run on hardware
//
#include <hip/hip_runtime.h>
#include <hip/hip_bf16.h>
#include <hip/hip_fp16.h>

// ---------------- problem constants ----------------
#define BATCH   8
#define MCH     12
#define ODIM    48
#define SLEN    1024
#define NDIM    64
#define DDIM    3072          // ODIM * NDIM
#define LN_EPS  1e-5f
#define ATT_SCALE 0.03608439182435161f   // 1/sqrt(768)

typedef __attribute__((ext_vector_type(16))) _Float16 v16h;
typedef __attribute__((ext_vector_type(8)))  _Float16 v8h;
typedef __attribute__((ext_vector_type(8)))  float    v8f;

union AFrag { v16h v; v8h h[2]; _Float16 e[16]; };

__device__ __forceinline__ float gelu_erf(float v) {
    return 0.5f * v * (1.0f + erff(v * 0.70710678118654752f));
}

// =====================================================================
// Kernel 1: grouped 3x3 conv (groups=12, 4 out-ch per in-ch) + bias +
// exact GELU for q,k,v.  Writes Q,K as f16 [B,S,D] (D = o*64+n) and
// V transposed as f16 [B,D,S] so the P·V B-fragments are contiguous.
// grid = (S/16, ODIM, B), block = 1024 (n = tid&63, ls = tid>>6)
// =====================================================================
__global__ __launch_bounds__(1024)
void qkv_conv_kernel(const float* __restrict__ x,
                     const float* __restrict__ wq, const float* __restrict__ bq,
                     const float* __restrict__ wk, const float* __restrict__ bk,
                     const float* __restrict__ wv, const float* __restrict__ bv,
                     _Float16* __restrict__ Q, _Float16* __restrict__ K,
                     _Float16* __restrict__ Vt)
{
    const int t  = threadIdx.x;
    const int n  = t & 63;
    const int ls = t >> 6;                 // 0..15
    const int o  = blockIdx.y;             // 0..47
    const int bb = blockIdx.z;             // 0..7
    const int s  = blockIdx.x * 16 + ls;   // 0..1023
    const int c  = o >> 2;                 // input channel (4 outputs per input)

    const float* xp = x + (((size_t)bb * MCH + c) * SLEN) * NDIM;

    float xv[9];
#pragma unroll
    for (int di = 0; di < 3; ++di) {
        const int ss = s + di - 1;
#pragma unroll
        for (int dj = 0; dj < 3; ++dj) {
            const int nn = n + dj - 1;
            const bool ok = (ss >= 0) && (ss < SLEN) && (nn >= 0) && (nn < NDIM);
            xv[di * 3 + dj] = ok ? xp[(size_t)ss * NDIM + nn] : 0.0f;
        }
    }

    float aq = bq[o], ak = bk[o], av = bv[o];
#pragma unroll
    for (int i = 0; i < 9; ++i) {
        aq = fmaf(wq[o * 9 + i], xv[i], aq);
        ak = fmaf(wk[o * 9 + i], xv[i], ak);
        av = fmaf(wv[o * 9 + i], xv[i], av);
    }
    aq = gelu_erf(aq); ak = gelu_erf(ak); av = gelu_erf(av);

    const size_t qi = ((size_t)bb * SLEN + s) * DDIM + (size_t)o * NDIM + n;
    Q[qi] = (_Float16)aq;
    K[qi] = (_Float16)ak;

    // transpose V through LDS: tile[n][s_local], then row-wise 32B stores
    __shared__ __align__(32) _Float16 vt[NDIM][16];
    vt[n][ls] = (_Float16)av;
    __syncthreads();
    if (t < NDIM) {
        const v16h row = *(const v16h*)(&vt[t][0]);
        *(v16h*)(Vt + ((size_t)bb * DDIM + (size_t)o * NDIM + t) * SLEN
                 + (size_t)blockIdx.x * 16) = row;
    }
}

// =====================================================================
// Kernel 2: attention.  One wave owns 16 query rows.
//   Phase 1: scores[16][1024] (f32, LDS) = Q Kᵀ via wmma_f32_16x16x32_f16,
//            4 K-tiles per Q A-fragment (40B cache traffic per WMMA)
//   Phase 2: exact softmax in LDS (scale, rowmax, exp; 1/Σ kept aside)
//   Phase 3: O[16, d-tile] = P·V via wmma, P read from LDS (f32→f16),
//            B-fragments from Vᵀ (contiguous), scale by 1/Σ at store.
// grid = (S/32, B), block = 64 (2 waves), dyn-LDS = 2*64KB + 128B
// =====================================================================
__global__ __launch_bounds__(64)
void attn_wmma_kernel(const _Float16* __restrict__ Q,
                      const _Float16* __restrict__ K,
                      const _Float16* __restrict__ Vt,
                      float* __restrict__ O)
{
    extern __shared__ float lds[];
    const int wave = threadIdx.x >> 5;
    const int lane = threadIdx.x & 31;
    const int bb   = blockIdx.y;
    const int r0   = blockIdx.x * 32 + wave * 16;

    float* sc   = lds + (size_t)wave * (16 * SLEN);          // 16 x 1024 f32
    float* invl = lds + 2 * 16 * SLEN + wave * 16;           // 16 f32

    const int row = lane & 15;
    const int hi  = lane >> 4;
    const int ncl = lane & 15;

    const _Float16* Qb = Q  + ((size_t)bb * SLEN) * DDIM;
    const _Float16* Kb = K  + ((size_t)bb * SLEN) * DDIM;
    const _Float16* Vb = Vt + ((size_t)bb * DDIM) * SLEN;

    // ---------------- Phase 1: raw scores -> LDS ----------------
    // A frag (16x32 f16): lane half holds K {hi*16..+15}
    const _Float16* qrow = Qb + (size_t)(r0 + row) * DDIM + hi * 16;
    for (int t0 = 0; t0 < SLEN; t0 += 64) {
        v8f c[4] = {{}, {}, {}, {}};
        // B frag (32x16 f16): lane holds 16 contiguous K of column (lane&15)
        const _Float16* kr = Kb + (size_t)(t0 + row) * DDIM + hi * 16;
        for (int dk = 0; dk < DDIM; dk += 32) {
            AFrag a;
            a.h[0] = *(const v8h*)(qrow + dk);
            a.h[1] = *(const v8h*)(qrow + dk + 8);
#pragma unroll
            for (int j = 0; j < 4; ++j) {
                const v16h bf = *(const v16h*)(kr + (size_t)(j * 16) * DDIM + dk);
                c[j] = __builtin_amdgcn_wmma_f32_16x16x32_f16(
                    false, a.v, false, bf, (short)0, c[j], false, false);
            }
        }
#pragma unroll
        for (int j = 0; j < 4; ++j)    // C layout: row v+8*hi, col lane&15
#pragma unroll
            for (int v = 0; v < 8; ++v)
                sc[(v + hi * 8) * SLEN + t0 + j * 16 + ncl] = c[j][v];
    }

    // ---------------- Phase 2: softmax (2 lanes per row) ----------------
    {
        float4* p4 = (float4*)(sc + (lane >> 1) * SLEN + (lane & 1) * 512);
        float mx = -3.4e38f;
        for (int i = 0; i < 128; ++i) {
            const float4 v = p4[i];
            mx = fmaxf(mx, fmaxf(fmaxf(v.x, v.y), fmaxf(v.z, v.w)));
        }
        mx = fmaxf(mx, __shfl_xor(mx, 1, 32));
        mx *= ATT_SCALE;
        float sum = 0.0f;
        for (int i = 0; i < 128; ++i) {
            float4 v = p4[i];
            v.x = __expf(v.x * ATT_SCALE - mx);
            v.y = __expf(v.y * ATT_SCALE - mx);
            v.z = __expf(v.z * ATT_SCALE - mx);
            v.w = __expf(v.w * ATT_SCALE - mx);
            p4[i] = v;
            sum += v.x + v.y + v.z + v.w;
        }
        sum += __shfl_xor(sum, 1, 32);
        if ((lane & 1) == 0) invl[lane >> 1] = 1.0f / sum;
    }

    // ---------------- Phase 3: O = P · V ----------------
    float inv[8];
#pragma unroll
    for (int v = 0; v < 8; ++v) inv[v] = invl[v + hi * 8];

    float* out = O + ((size_t)bb * SLEN + r0) * DDIM;
    for (int d0 = 0; d0 < DDIM; d0 += 64) {
        v8f acc[4] = {{}, {}, {}, {}};
        for (int t0 = 0; t0 < SLEN; t0 += 32) {
            // A frag from LDS probabilities (f32 -> f16)
            const float* pa = sc + row * SLEN + t0 + hi * 16;
            const float4 f0 = *(const float4*)(pa);
            const float4 f1 = *(const float4*)(pa + 4);
            const float4 f2 = *(const float4*)(pa + 8);
            const float4 f3 = *(const float4*)(pa + 12);
            AFrag a;
            a.e[0]  = (_Float16)f0.x; a.e[1]  = (_Float16)f0.y;
            a.e[2]  = (_Float16)f0.z; a.e[3]  = (_Float16)f0.w;
            a.e[4]  = (_Float16)f1.x; a.e[5]  = (_Float16)f1.y;
            a.e[6]  = (_Float16)f1.z; a.e[7]  = (_Float16)f1.w;
            a.e[8]  = (_Float16)f2.x; a.e[9]  = (_Float16)f2.y;
            a.e[10] = (_Float16)f2.z; a.e[11] = (_Float16)f2.w;
            a.e[12] = (_Float16)f3.x; a.e[13] = (_Float16)f3.y;
            a.e[14] = (_Float16)f3.z; a.e[15] = (_Float16)f3.w;
#pragma unroll
            for (int j = 0; j < 4; ++j) {
                const v16h bfrag = *(const v16h*)(Vb
                    + (size_t)(d0 + j * 16 + row) * SLEN + t0 + hi * 16);
                acc[j] = __builtin_amdgcn_wmma_f32_16x16x32_f16(
                    false, a.v, false, bfrag, (short)0, acc[j], false, false);
            }
        }
#pragma unroll
        for (int j = 0; j < 4; ++j)
#pragma unroll
            for (int v = 0; v < 8; ++v)
                out[(size_t)(v + hi * 8) * DDIM + d0 + j * 16 + ncl] = acc[j][v] * inv[v];
    }
}

// =====================================================================
// Kernel 3: 1x1 conv (48 -> 12) + bias + LayerNorm over N=64 + residual.
// grid = (S, B), block = 64 (one lane per n)
// =====================================================================
__global__ __launch_bounds__(64)
void epilogue_kernel(const float* __restrict__ O, const float* __restrict__ x,
                     const float* __restrict__ w1, const float* __restrict__ b1,
                     const float* __restrict__ lnw, const float* __restrict__ lnb,
                     float* __restrict__ out)
{
    const int n  = threadIdx.x;
    const int s  = blockIdx.x;
    const int bb = blockIdx.y;
    const float* op = O + ((size_t)bb * SLEN + s) * DDIM + n;

    float acc[MCH];
#pragma unroll
    for (int m = 0; m < MCH; ++m) acc[m] = b1[m];
#pragma unroll 4
    for (int o = 0; o < ODIM; ++o) {
        const float v = op[(size_t)o * NDIM];
#pragma unroll
        for (int m = 0; m < MCH; ++m) acc[m] = fmaf(w1[m * ODIM + o], v, acc[m]);
    }

    __shared__ float s1[MCH * NDIM];
    __shared__ float s2[MCH * NDIM];
#pragma unroll
    for (int m = 0; m < MCH; ++m) {
        s1[m * NDIM + n] = acc[m];
        s2[m * NDIM + n] = acc[m] * acc[m];
    }
    __syncthreads();
    for (int st = 32; st > 0; st >>= 1) {
        if (n < st) {
#pragma unroll
            for (int m = 0; m < MCH; ++m) {
                s1[m * NDIM + n] += s1[m * NDIM + n + st];
                s2[m * NDIM + n] += s2[m * NDIM + n + st];
            }
        }
        __syncthreads();
    }
    const float lw = lnw[n], lb = lnb[n];
#pragma unroll
    for (int m = 0; m < MCH; ++m) {
        const float mu  = s1[m * NDIM] * (1.0f / NDIM);
        const float var = s2[m * NDIM] * (1.0f / NDIM) - mu * mu;
        const float r   = rsqrtf(var + LN_EPS);
        const size_t idx = (((size_t)bb * MCH + m) * SLEN + s) * NDIM + n;
        out[idx] = (acc[m] - mu) * r * lw + lb + x[idx];
    }
}

// =====================================================================
extern "C" void kernel_launch(void* const* d_in, const int* in_sizes, int n_in,
                              void* d_out, int out_size, void* d_ws, size_t ws_size,
                              hipStream_t stream) {
    (void)in_sizes; (void)n_in; (void)out_size; (void)ws_size;
    const float* x   = (const float*)d_in[0];
    const float* wq  = (const float*)d_in[1];
    const float* bq  = (const float*)d_in[2];
    const float* wk  = (const float*)d_in[3];
    const float* bk  = (const float*)d_in[4];
    const float* wv  = (const float*)d_in[5];
    const float* bv  = (const float*)d_in[6];
    const float* w1  = (const float*)d_in[7];
    const float* b1  = (const float*)d_in[8];
    const float* lnw = (const float*)d_in[9];
    const float* lnb = (const float*)d_in[10];

    const size_t E = (size_t)BATCH * SLEN * DDIM;   // 25,165,824
    _Float16* Q  = (_Float16*)d_ws;
    _Float16* K  = Q + E;
    _Float16* Vt = K + E;
    float*    O  = (float*)(Vt + E);

    dim3 g1(SLEN / 16, ODIM, BATCH);
    qkv_conv_kernel<<<g1, 1024, 0, stream>>>(x, wq, bq, wk, bk, wv, bv, Q, K, Vt);

    dim3 g2(SLEN / 32, BATCH);
    const size_t smem = 2 * 16 * SLEN * sizeof(float) + 2 * 16 * sizeof(float);
    attn_wmma_kernel<<<g2, 64, smem, stream>>>(Q, K, Vt, O);

    dim3 g3(SLEN, BATCH);
    epilogue_kernel<<<g3, 64, 0, stream>>>(O, x, w1, b1, lnw, lnb, (float*)d_out);
}
